// HybridLocalAggregator_29248727286399
// MI455X (gfx1250) — matrix operand order repaired and checked by reading, not compiled, not run on hardware
//
#include <hip/hip_runtime.h>

// ---------- types / helpers ----------
typedef __attribute__((ext_vector_type(16))) __bf16 vbf;
typedef __attribute__((ext_vector_type(8)))  __bf16 vbf8;
typedef __attribute__((ext_vector_type(8)))  float  v8f;
typedef __attribute__((ext_vector_type(8)))  unsigned short ushort8;

__device__ __forceinline__ unsigned short f2bf(float f) {
    unsigned u = __builtin_bit_cast(unsigned, f);
    unsigned r = u + 0x7FFFu + ((u >> 16) & 1u);   // round-to-nearest-even
    return (unsigned short)(r >> 16);
}
__device__ __forceinline__ float bf2f(unsigned short h) {
    unsigned u = ((unsigned)h) << 16;
    return __builtin_bit_cast(float, u);
}
// order-preserving float<->uint encoding for atomicMax
__device__ __forceinline__ unsigned fenc(float f) {
    unsigned u = __builtin_bit_cast(unsigned, f);
    return (u & 0x80000000u) ? ~u : (u | 0x80000000u);
}
__device__ __forceinline__ float fdec(unsigned u) {
    unsigned v = (u & 0x80000000u) ? (u & 0x7FFFFFFFu) : ~u;
    return __builtin_bit_cast(float, v);
}
__device__ __forceinline__ bool finitef(float f) {
    unsigned u = __builtin_bit_cast(unsigned, f);
    return ((u >> 23) & 0xFFu) != 0xFFu;
}
#define SHUF16(lo, hi) __builtin_shufflevector(lo, hi, 0,1,2,3,4,5,6,7,8,9,10,11,12,13,14,15)

// ---------- BN stats over x [N,64] ----------
__global__ void k_xstats(const float* __restrict__ x, float* xsum, float* xsq, int N) {
    int g = blockIdx.x * blockDim.x + threadIdx.x;
    int c = g & 63;
    int r = g >> 6;
    int rs = (gridDim.x * blockDim.x) >> 6;
    float s = 0.f, q = 0.f;
    for (; r < N; r += rs) {
        float v = x[(size_t)r * 64 + c];
        s += v; q += v * v;
    }
    atomicAdd(&xsum[c], s);
    atomicAdd(&xsq[c], q);
}

// finalize BN: a = g*rsqrt(var+eps), c = b - mean*a   (64 channels)
__global__ void k_finalize(const float* __restrict__ sum, const float* __restrict__ sq,
                           const float* __restrict__ g, const float* __restrict__ b,
                           float cnt, float* a, float* cvec) {
    int c = threadIdx.x;
    if (c < 64) {
        float m = sum[c] / cnt;
        float v = sq[c] / cnt - m * m;
        float aa = g[c] * rsqrtf(v + 1e-5f);
        a[c] = aa;
        cvec[c] = b[c] - m * aa;
    }
}

// h = a0[c]*x + c0[c]  (float4 over N*16 vectors)
__global__ void k_apply_h(const float4* __restrict__ x4, const float* __restrict__ a0,
                          const float* __restrict__ c0, float4* __restrict__ h4, size_t total4) {
    size_t i = (size_t)blockIdx.x * blockDim.x + threadIdx.x;
    if (i < total4) {
        int cb = (int)((i & 15) << 2);
        float4 v = x4[i];
        float4 r;
        r.x = a0[cb + 0] * v.x + c0[cb + 0];
        r.y = a0[cb + 1] * v.y + c0[cb + 1];
        r.z = a0[cb + 2] * v.z + c0[cb + 2];
        r.w = a0[cb + 3] * v.w + c0[cb + 3];
        h4[i] = r;
    }
}

// pack a [K,64] f32 weight matrix into WMMA B-fragment-major bf16 layout:
// frag f = (kk*4+nt)*32+lane ; elements j=0..15 contiguous ; K in {64,128}
__global__ void k_packW(const float* __restrict__ W, unsigned short* __restrict__ Wp, int K) {
    int f = blockIdx.x * blockDim.x + threadIdx.x;
    int total = (K >> 5) * 128;
    if (f >= total) return;
    int lane = f & 31, nt = (f >> 5) & 3, kk = f >> 7;
    int n = nt * 16 + (lane & 15);
    int kbase = kk * 32 + ((lane >> 4) << 3);
    unsigned short tmp[16];
#pragma unroll
    for (int j = 0; j < 16; ++j) tmp[j] = f2bf(W[(kbase + (j & 7) + ((j >> 3) << 4)) * 64 + n]);
    ushort8* dst = (ushort8*)(Wp + (size_t)f * 16);
    dst[0] = *(const ushort8*)&tmp[0];
    dst[1] = *(const ushort8*)&tmp[8];
}

// ---------- edge GEMM1: z1 = relu([h_row, h_row-h_col] @ W1 + b1), bf16 out + BN1 stats ----------
__global__ __launch_bounds__(256) void k_gemm1(const float* __restrict__ h,
                                               const int* __restrict__ rowI,
                                               const int* __restrict__ colI,
                                               const unsigned short* __restrict__ Wp,
                                               const float* __restrict__ b1,
                                               unsigned short* __restrict__ z,
                                               float* s1sum, float* s1sq, int E) {
    __shared__ __align__(16) unsigned short sA[8 * 16 * 128];
    __shared__ float ssum[64], ssq[64];
    int tid = threadIdx.x;
    if (tid < 64) { ssum[tid] = 0.f; ssq[tid] = 0.f; }
    int wave = tid >> 5, lane = tid & 31;
    int e0 = blockIdx.x * 128 + wave * 16;
    unsigned short* A = sA + wave * (16 * 128);
    for (int m = 0; m < 16; ++m) {
        int e = e0 + m;
        int ee = e < E ? e : (E - 1);
        int r = rowI[ee], c = colI[ee];
        float2 av = *(const float2*)(h + (size_t)r * 64 + lane * 2);
        float2 bv = *(const float2*)(h + (size_t)c * 64 + lane * 2);
        unsigned lo = (unsigned)f2bf(av.x) | ((unsigned)f2bf(av.y) << 16);
        unsigned df = (unsigned)f2bf(av.x - bv.x) | ((unsigned)f2bf(av.y - bv.y) << 16);
        *(unsigned*)&A[m * 128 + lane * 2]      = lo;
        *(unsigned*)&A[m * 128 + 64 + lane * 2] = df;
    }
    __syncthreads();

    int mhalf = lane >> 4, l15 = lane & 15, kb = mhalf * 8;
    v8f acc[4] = {};
#pragma unroll
    for (int kk = 0; kk < 4; ++kk) {
        vbf8 alo = *(const vbf8*)(A + l15 * 128 + kk * 32 + kb);
        vbf8 ahi = *(const vbf8*)(A + l15 * 128 + kk * 32 + 16 + kb);
        vbf a = SHUF16(alo, ahi);
#pragma unroll
        for (int nt = 0; nt < 4; ++nt) {
            const vbf8* bp = (const vbf8*)(Wp + (size_t)(((kk * 4 + nt) * 32 + lane)) * 16);
            vbf b = SHUF16(bp[0], bp[1]);
            acc[nt] = __builtin_amdgcn_wmma_f32_16x16x32_bf16(false, a, false, b,
                                                              (short)0, acc[nt], false, false);
        }
    }
    __syncthreads();          // all waves done with A fragments; reuse region as output tile
    unsigned short* O = A;    // 16 x 64 bf16
#pragma unroll
    for (int nt = 0; nt < 4; ++nt) {
        int n = nt * 16 + l15;
        float bb = b1[n];
        float ls = 0.f, lq = 0.f;
#pragma unroll
        for (int v = 0; v < 8; ++v) {
            int e = e0 + mhalf * 8 + v;
            float val = fmaxf(acc[nt][v] + bb, 0.f);
            O[(mhalf * 8 + v) * 64 + n] = f2bf(val);
            if (e < E) { ls += val; lq += val * val; }
        }
        atomicAdd(&ssum[n], ls);
        atomicAdd(&ssq[n], lq);
    }
    __syncthreads();
    // coalesced b128 stores: 128 chunks of 8 bf16 per wave tile
    for (int ci = lane; ci < 128; ci += 32) {
        int r = ci >> 3, cc = ci & 7;
        int e = e0 + r;
        if (e < E)
            *(ushort8*)(z + (size_t)e * 64 + cc * 8) = *(const ushort8*)(O + r * 64 + cc * 8);
    }
    if (tid < 64) { atomicAdd(&s1sum[tid], ssum[tid]); atomicAdd(&s1sq[tid], ssq[tid]); }
}

// fold BN1 affine into W2:  W2s[k][n] = a1[k]*W2[k][n] (f32),  b2'[n] = b2[n] + sum_k c1[k]*W2[k][n]
__global__ void k_foldW2(const float* a1, const float* c1, const float* __restrict__ W2,
                         const float* __restrict__ b2, float* W2s, float* b2p) {
    int n = threadIdx.x;
    if (n < 64) {
        float acc = b2[n];
        for (int k = 0; k < 64; ++k) {
            float w = W2[k * 64 + n];
            W2s[k * 64 + n] = a1[k] * w;
            acc += c1[k] * w;
        }
        b2p[n] = acc;
    }
}

// ---------- edge GEMM2 (in place): z = relu(z @ W2' + b2') + BN2 stats ----------
__global__ __launch_bounds__(256) void k_gemm2(unsigned short* __restrict__ z,
                                               const unsigned short* __restrict__ Wp,
                                               const float* __restrict__ b2p,
                                               float* s2sum, float* s2sq, int E) {
    __shared__ __align__(16) unsigned short sO[8 * 16 * 64];
    __shared__ float ssum[64], ssq[64];
    int tid = threadIdx.x;
    if (tid < 64) { ssum[tid] = 0.f; ssq[tid] = 0.f; }
    __syncthreads();
    int wave = tid >> 5, lane = tid & 31;
    int e0 = blockIdx.x * 128 + wave * 16;
    int mhalf = lane >> 4, l15 = lane & 15, kb = mhalf * 8;
    int em = e0 + l15;
    int emc = em < E ? em : (E - 1);
    const unsigned short* zr = z + (size_t)emc * 64;
    v8f acc[4] = {};
#pragma unroll
    for (int kk = 0; kk < 2; ++kk) {
        vbf8 alo = *(const vbf8*)(zr + kk * 32 + kb);
        vbf8 ahi = *(const vbf8*)(zr + kk * 32 + 16 + kb);
        vbf a = SHUF16(alo, ahi);
#pragma unroll
        for (int nt = 0; nt < 4; ++nt) {
            const vbf8* bp = (const vbf8*)(Wp + (size_t)(((kk * 4 + nt) * 32 + lane)) * 16);
            vbf b = SHUF16(bp[0], bp[1]);
            acc[nt] = __builtin_amdgcn_wmma_f32_16x16x32_bf16(false, a, false, b,
                                                              (short)0, acc[nt], false, false);
        }
    }
    unsigned short* O = sO + wave * (16 * 64);
#pragma unroll
    for (int nt = 0; nt < 4; ++nt) {
        int n = nt * 16 + l15;
        float bb = b2p[n];
        float ls = 0.f, lq = 0.f;
#pragma unroll
        for (int v = 0; v < 8; ++v) {
            int e = e0 + mhalf * 8 + v;
            float val = fmaxf(acc[nt][v] + bb, 0.f);
            O[(mhalf * 8 + v) * 64 + n] = f2bf(val);
            if (e < E) { ls += val; lq += val * val; }
        }
        atomicAdd(&ssum[n], ls);
        atomicAdd(&ssq[n], lq);
    }
    __syncthreads();
    for (int ci = lane; ci < 128; ci += 32) {
        int r = ci >> 3, cc = ci & 7;
        int e = e0 + r;
        if (e < E)
            *(ushort8*)(z + (size_t)e * 64 + cc * 8) = *(const ushort8*)(O + r * 64 + cc * 8);
    }
    if (tid < 64) { atomicAdd(&s2sum[tid], ssum[tid]); atomicAdd(&s2sq[tid], ssq[tid]); }
}

// fold BN2 into attention vector: attw2 = a2*attW; attb2 = attb + dot(c2, attW)
__global__ void k_foldatt(const float* a2, const float* c2, const float* __restrict__ attW,
                          const float* __restrict__ attb, float* attw2, float* attb2) {
    __shared__ float red[64];
    int c = threadIdx.x;
    float w = attW[c];
    attw2[c] = a2[c] * w;
    red[c] = c2[c] * w;
    __syncthreads();
    if (c == 0) {
        float s = attb[0];
        for (int i = 0; i < 64; ++i) s += red[i];
        attb2[0] = s;
    }
}

// per-edge: logits, segment-max of logits, channel-wise segment-max of e = z*a2+c2
__global__ __launch_bounds__(256) void k_logit_max(const unsigned short* __restrict__ z,
                                                   const int* __restrict__ colI,
                                                   const float* a2, const float* c2,
                                                   const float* attw2, const float* attb2,
                                                   float* __restrict__ logits,
                                                   unsigned* __restrict__ nlm,
                                                   unsigned* __restrict__ omax, int E) {
    __shared__ float sa[64], sc[64], sw[64];
    int tid = threadIdx.x;
    if (tid < 64) { sa[tid] = a2[tid]; sc[tid] = c2[tid]; sw[tid] = attw2[tid]; }
    __syncthreads();
    int e = blockIdx.x * 256 + tid;
    if (e >= E) return;
    int cn = colI[e];
    const ushort8* zr8 = (const ushort8*)(z + (size_t)e * 64);
    unsigned* om = omax + (size_t)cn * 64;
    float lg = attb2[0];
#pragma unroll
    for (int c8 = 0; c8 < 8; ++c8) {
        ushort8 v8 = zr8[c8];
#pragma unroll
        for (int j = 0; j < 8; ++j) {
            int c = c8 * 8 + j;
            float zv = bf2f(v8[j]);
            lg += zv * sw[c];
            atomicMax(&om[c], fenc(zv * sa[c] + sc[c]));
        }
    }
    logits[e] = lg;
    atomicMax(&nlm[cn], fenc(lg));
}

// per-edge: ex = exp(logit - segmax); den[col]+=ex; aatt[col][c] += ex*e_c
__global__ __launch_bounds__(256) void k_att(const unsigned short* __restrict__ z,
                                             const int* __restrict__ colI,
                                             const float* a2, const float* c2,
                                             const float* __restrict__ logits,
                                             const unsigned* __restrict__ nlm,
                                             float* __restrict__ den,
                                             float* __restrict__ aatt, int E) {
    __shared__ float sa[64], sc[64];
    int tid = threadIdx.x;
    if (tid < 64) { sa[tid] = a2[tid]; sc[tid] = c2[tid]; }
    __syncthreads();
    int e = blockIdx.x * 256 + tid;
    if (e >= E) return;
    int cn = colI[e];
    float mx = fdec(nlm[cn]);
    float ex = __expf(logits[e] - mx);
    atomicAdd(&den[cn], ex);
    const ushort8* zr8 = (const ushort8*)(z + (size_t)e * 64);
    float* ac = aatt + (size_t)cn * 64;
#pragma unroll
    for (int c8 = 0; c8 < 8; ++c8) {
        ushort8 v8 = zr8[c8];
#pragma unroll
        for (int j = 0; j < 8; ++j) {
            int c = c8 * 8 + j;
            atomicAdd(&ac[c], ex * (bf2f(v8[j]) * sa[c] + sc[c]));
        }
    }
}

// ---------- fusion GEMM: out_pre = [out_max, out_att] @ fus_W + fus_b  + out-BN stats ----------
__global__ __launch_bounds__(256) void k_fusion(const unsigned* __restrict__ omax,
                                                const float* __restrict__ aatt,
                                                const float* __restrict__ den,
                                                const unsigned short* __restrict__ Wp,
                                                const float* __restrict__ fusb,
                                                float* __restrict__ opre,
                                                float* osum, float* osq, int N) {
    __shared__ __align__(16) unsigned short sA[8 * 16 * 128];
    __shared__ float ssum[64], ssq[64];
    int tid = threadIdx.x;
    if (tid < 64) { ssum[tid] = 0.f; ssq[tid] = 0.f; }
    int wave = tid >> 5, lane = tid & 31;
    int n0 = blockIdx.x * 128 + wave * 16;
    unsigned short* A = sA + wave * (16 * 128);
    for (int m = 0; m < 16; ++m) {
        int nd = n0 + m;
        int ndc = nd < N ? nd : (N - 1);
        float dv = den[ndc];
        float inv = (dv > 0.f) ? (1.f / dv) : 0.f;
        uint2 u2 = *(const uint2*)(omax + (size_t)ndc * 64 + lane * 2);
        float2 at = *(const float2*)(aatt + (size_t)ndc * 64 + lane * 2);
        float m0 = 0.f, m1 = 0.f;
        if (u2.x) { float d = fdec(u2.x); if (finitef(d)) m0 = d; }
        if (u2.y) { float d = fdec(u2.y); if (finitef(d)) m1 = d; }
        *(unsigned*)&A[m * 128 + lane * 2] =
            (unsigned)f2bf(m0) | ((unsigned)f2bf(m1) << 16);
        *(unsigned*)&A[m * 128 + 64 + lane * 2] =
            (unsigned)f2bf(at.x * inv) | ((unsigned)f2bf(at.y * inv) << 16);
    }
    __syncthreads();

    int mhalf = lane >> 4, l15 = lane & 15, kb = mhalf * 8;
    v8f acc[4] = {};
#pragma unroll
    for (int kk = 0; kk < 4; ++kk) {
        vbf8 alo = *(const vbf8*)(A + l15 * 128 + kk * 32 + kb);
        vbf8 ahi = *(const vbf8*)(A + l15 * 128 + kk * 32 + 16 + kb);
        vbf a = SHUF16(alo, ahi);
#pragma unroll
        for (int nt = 0; nt < 4; ++nt) {
            const vbf8* bp = (const vbf8*)(Wp + (size_t)(((kk * 4 + nt) * 32 + lane)) * 16);
            vbf b = SHUF16(bp[0], bp[1]);
            acc[nt] = __builtin_amdgcn_wmma_f32_16x16x32_bf16(false, a, false, b,
                                                              (short)0, acc[nt], false, false);
        }
    }
    __syncthreads();          // reuse A region (4KB) as f32 output tile (16x64)
    float* Of = (float*)A;
#pragma unroll
    for (int nt = 0; nt < 4; ++nt) {
        int n = nt * 16 + l15;
        float bb = fusb[n];
        float ls = 0.f, lq = 0.f;
#pragma unroll
        for (int v = 0; v < 8; ++v) {
            int nd = n0 + mhalf * 8 + v;
            float val = acc[nt][v] + bb;        // no relu before final BN
            Of[(mhalf * 8 + v) * 64 + n] = val;
            if (nd < N) { ls += val; lq += val * val; }
        }
        atomicAdd(&ssum[n], ls);
        atomicAdd(&ssq[n], lq);
    }
    __syncthreads();
    // coalesced float4 stores: 256 chunks per wave tile
    for (int ci = lane; ci < 256; ci += 32) {
        int r = ci >> 4, cc = ci & 15;
        int nd = n0 + r;
        if (nd < N)
            *(float4*)(opre + (size_t)nd * 64 + cc * 4) = *(const float4*)(Of + r * 64 + cc * 4);
    }
    if (tid < 64) { atomicAdd(&osum[tid], ssum[tid]); atomicAdd(&osq[tid], ssq[tid]); }
}

// final: out = relu(a3[c]*out_pre + c3[c])   (float4)
__global__ void k_final(const float4* __restrict__ opre4, const float* a3, const float* c3,
                        float4* __restrict__ out4, size_t total4) {
    size_t i = (size_t)blockIdx.x * blockDim.x + threadIdx.x;
    if (i < total4) {
        int cb = (int)((i & 15) << 2);
        float4 v = opre4[i];
        float4 r;
        r.x = fmaxf(v.x * a3[cb + 0] + c3[cb + 0], 0.f);
        r.y = fmaxf(v.y * a3[cb + 1] + c3[cb + 1], 0.f);
        r.z = fmaxf(v.z * a3[cb + 2] + c3[cb + 2], 0.f);
        r.w = fmaxf(v.w * a3[cb + 3] + c3[cb + 3], 0.f);
        out4[i] = r;
    }
}

// ---------- launch ----------
extern "C" void kernel_launch(void* const* d_in, const int* in_sizes, int n_in,
                              void* d_out, int out_size, void* d_ws, size_t ws_size,
                              hipStream_t stream) {
    (void)n_in; (void)out_size; (void)ws_size;
    const float* x     = (const float*)d_in[0];
    const float* in_g  = (const float*)d_in[1];
    const float* in_b  = (const float*)d_in[2];
    const float* W1    = (const float*)d_in[3];
    const float* b1    = (const float*)d_in[4];
    const float* bn1_g = (const float*)d_in[5];
    const float* bn1_b = (const float*)d_in[6];
    const float* W2    = (const float*)d_in[7];
    const float* b2    = (const float*)d_in[8];
    const float* bn2_g = (const float*)d_in[9];
    const float* bn2_b = (const float*)d_in[10];
    const float* attW  = (const float*)d_in[11];
    const float* attb  = (const float*)d_in[12];
    const float* fusW  = (const float*)d_in[13];
    const float* fusb  = (const float*)d_in[14];
    const float* out_g = (const float*)d_in[15];
    const float* out_b = (const float*)d_in[16];
    const int*   eidx  = (const int*)d_in[17];

    int N = in_sizes[0] / 64;
    int E = in_sizes[17] / 2;
    const int* rowI = eidx;
    const int* colI = eidx + E;

    char* ws = (char*)d_ws;
    size_t off = 0;
    auto take = [&](size_t bytes) -> char* {
        char* p = ws + off;
        off = (off + bytes + 255) & ~(size_t)255;
        return p;
    };
    float* stats = (float*)take(512 * sizeof(float));
    float *xsum = stats, *xsq = stats + 64, *s1sum = stats + 128, *s1sq = stats + 192;
    float *s2sum = stats + 256, *s2sq = stats + 320, *osum = stats + 384, *osq = stats + 448;
    float* a0 = (float*)take(64 * 4);  float* c0 = (float*)take(64 * 4);
    float* a1 = (float*)take(64 * 4);  float* c1 = (float*)take(64 * 4);
    float* a2 = (float*)take(64 * 4);  float* c2 = (float*)take(64 * 4);
    float* a3 = (float*)take(64 * 4);  float* c3 = (float*)take(64 * 4);
    float* attw2 = (float*)take(64 * 4);
    float* attb2 = (float*)take(256);
    float* b2p   = (float*)take(64 * 4);
    float* W2s   = (float*)take(64 * 64 * 4);
    unsigned short* W1p  = (unsigned short*)take(512 * 16 * 2);   // 4 ksteps * 128 frag-lanes
    unsigned short* W2p  = (unsigned short*)take(256 * 16 * 2);   // 2 ksteps
    unsigned short* fusp = (unsigned short*)take(512 * 16 * 2);
    float* h            = (float*)take((size_t)N * 64 * 4);
    unsigned short* z   = (unsigned short*)take((size_t)E * 64 * 2);
    float* logits       = (float*)take((size_t)E * 4);
    unsigned* nlm       = (unsigned*)take((size_t)N * 4);
    float* den          = (float*)take((size_t)N * 4);
    unsigned* omax      = (unsigned*)take((size_t)N * 64 * 4);
    float* aatt         = (float*)take((size_t)N * 64 * 4);
    float* opre         = (float*)take((size_t)N * 64 * 4);

    // per-call init of all accumulated scratch (graph-capturable)
    hipMemsetAsync(stats, 0, 512 * sizeof(float), stream);
    hipMemsetAsync(nlm,  0, (size_t)N * 4, stream);
    hipMemsetAsync(den,  0, (size_t)N * 4, stream);
    hipMemsetAsync(omax, 0, (size_t)N * 64 * 4, stream);
    hipMemsetAsync(aatt, 0, (size_t)N * 64 * 4, stream);

    size_t totN4 = (size_t)N * 16;      // float4 count of N*64
    int ebl = (E + 127) / 128;
    int eb2 = (E + 255) / 256;
    int nbl = (N + 127) / 128;

    k_xstats  <<<256, 256, 0, stream>>>(x, xsum, xsq, N);
    k_finalize<<<1, 64, 0, stream>>>(xsum, xsq, in_g, in_b, (float)N, a0, c0);
    k_apply_h <<<(int)((totN4 + 255) / 256), 256, 0, stream>>>((const float4*)x, a0, c0,
                                                               (float4*)h, totN4);
    k_packW   <<<2, 256, 0, stream>>>(W1, W1p, 128);
    k_packW   <<<2, 256, 0, stream>>>(fusW, fusp, 128);

    k_gemm1   <<<ebl, 256, 0, stream>>>(h, rowI, colI, W1p, b1, z, s1sum, s1sq, E);
    k_finalize<<<1, 64, 0, stream>>>(s1sum, s1sq, bn1_g, bn1_b, (float)E, a1, c1);
    k_foldW2  <<<1, 64, 0, stream>>>(a1, c1, W2, b2, W2s, b2p);
    k_packW   <<<1, 256, 0, stream>>>(W2s, W2p, 64);

    k_gemm2   <<<ebl, 256, 0, stream>>>(z, W2p, b2p, s2sum, s2sq, E);
    k_finalize<<<1, 64, 0, stream>>>(s2sum, s2sq, bn2_g, bn2_b, (float)E, a2, c2);
    k_foldatt <<<1, 64, 0, stream>>>(a2, c2, attW, attb, attw2, attb2);

    k_logit_max<<<eb2, 256, 0, stream>>>(z, colI, a2, c2, attw2, attb2, logits, nlm, omax, E);
    k_att      <<<eb2, 256, 0, stream>>>(z, colI, a2, c2, logits, nlm, den, aatt, E);

    k_fusion  <<<nbl, 256, 0, stream>>>(omax, aatt, den, fusp, fusb, opre, osum, osq, N);
    k_finalize<<<1, 64, 0, stream>>>(osum, osq, out_g, out_b, (float)N, a3, c3);
    k_final   <<<(int)((totN4 + 255) / 256), 256, 0, stream>>>((const float4*)opre, a3, c3,
                                                               (float4*)d_out, totN4);
}
